// DecoderAttention_17944373363293
// MI455X (gfx1250) — compile-verified, matrix operand order checked by reading, not run on hardware
//
#include <hip/hip_runtime.h>
#include <hip/hip_bf16.h>

// ---------------------------------------------------------------------------
// DecoderAttention (GRU + input attention), MI455X / gfx1250, wave32.
// bf16 WMMA (v_wmma_f32_16x16x32_bf16) for all GEMMs, fp32 state & gates.
// GEMM inner loop is software double-buffered so fragment loads for slice k+1
// are in flight while the WMMAs for slice k execute.
// ---------------------------------------------------------------------------

typedef __attribute__((ext_vector_type(16))) __bf16 v16bf;
typedef __attribute__((ext_vector_type(8)))  __bf16 v8bf;
typedef __attribute__((ext_vector_type(8)))  float  v8f;

#define CAT16(lo, hi) __builtin_shufflevector(lo, hi, 0,1,2,3,4,5,6,7,8,9,10,11,12,13,14,15)

#define B_   256
#define T_   48
#define H_   512
#define G3_  1536      // 3*H
#define K_AH 4

// ---------------------------- prep kernels ---------------------------------

__global__ void k_convert_bf16(const float* __restrict__ in,
                               __bf16* __restrict__ out, int n) {
  int i = blockIdx.x * blockDim.x + threadIdx.x;
  if (i < n) out[i] = (__bf16)in[i];
}

// in: [K,N] row-major fp32  ->  out: [N,K] row-major bf16
__global__ void k_transpose_bf16(const float* __restrict__ in,
                                 __bf16* __restrict__ out, int K, int N) {
  int i = blockIdx.x * blockDim.x + threadIdx.x;
  if (i < K * N) {
    int n = i / K, k = i - n * K;
    out[i] = (__bf16)in[(size_t)k * N + n];
  }
}

// W_ih: [3H, S+1] -> bf16 [3H, S] (K=512 clean for WMMA) + fp32 last column
__global__ void k_split_wih(const float* __restrict__ Wih,
                            __bf16* __restrict__ WihS,
                            float* __restrict__ wlast) {
  int n = blockIdx.x;
  const float* row = Wih + (size_t)n * (H_ + 1);
  for (int k = threadIdx.x; k < H_; k += blockDim.x)
    WihS[(size_t)n * H_ + k] = (__bf16)row[k];
  if (threadIdx.x == 0) wlast[n] = row[H_];
}

__global__ void k_zero_state(float* __restrict__ d, __bf16* __restrict__ dbf, int n) {
  int i = blockIdx.x * blockDim.x + threadIdx.x;
  if (i < n) { d[i] = 0.0f; dbf[i] = (__bf16)0.0f; }
}

// ---------------------------- WMMA GEMM ------------------------------------
// C[M,N] = A[M,K] @ Bt[N,K]^T (+ bias[N]).  A,Bt bf16 row-major, C fp32.
// Block = 256 threads = 8 waves tiled 2x4; each wave owns a 16x32 C strip
// (two 16x16 WMMAs sharing one A fragment) -> 32x128 C tile per block.
// M = gridDim.x*32, N = gridDim.y*128, K multiple of 32; no guards so EXEC
// stays all-ones around v_wmma as the ISA requires.
__global__ void k_wmma_gemm(const __bf16* __restrict__ A,
                            const __bf16* __restrict__ Bt,
                            const float* __restrict__ bias,
                            float* __restrict__ C,
                            int N, int K) {
  const int lane = threadIdx.x & 31;
  const int wave = threadIdx.x >> 5;
  const int m0 = blockIdx.x * 32 + (wave >> 2) * 16;
  const int n0 = blockIdx.y * 128 + (wave & 3) * 32;

  // ISA 16-bit A layout: lanes 0-15 row M=lane, K=0..7 & 16..23;
  // lanes 16-31 same rows, K=8..15 & 24..31.  Bt uses identical pattern on N.
  const int koff = (lane >> 4) << 3;              // 0 or 8
  const __bf16* pa  = A  + (size_t)(m0 + (lane & 15)) * K + koff;
  const __bf16* pb0 = Bt + (size_t)(n0 + (lane & 15)) * K + koff;
  const __bf16* pb1 = pb0 + (size_t)16 * K;

  // prologue: load K-slice 0
  v8bf alo  = *(const v8bf*)(pa);
  v8bf ahi  = *(const v8bf*)(pa + 16);
  v8bf b0lo = *(const v8bf*)(pb0);
  v8bf b0hi = *(const v8bf*)(pb0 + 16);
  v8bf b1lo = *(const v8bf*)(pb1);
  v8bf b1hi = *(const v8bf*)(pb1 + 16);

  v8f acc0 = {}, acc1 = {};
  const int kiters = K >> 5;

  for (int it = 0; it < kiters - 1; ++it) {
    pa += 32; pb0 += 32; pb1 += 32;
    if (it + 2 < kiters) {                        // 2 slices ahead (L2-resident)
      __builtin_prefetch(pa + 32, 0, 2);
      __builtin_prefetch(pb0 + 32, 0, 2);
      __builtin_prefetch(pb1 + 32, 0, 2);
    }
    // issue next-slice loads before consuming current fragments
    v8bf nalo  = *(const v8bf*)(pa);
    v8bf nahi  = *(const v8bf*)(pa + 16);
    v8bf nb0lo = *(const v8bf*)(pb0);
    v8bf nb0hi = *(const v8bf*)(pb0 + 16);
    v8bf nb1lo = *(const v8bf*)(pb1);
    v8bf nb1hi = *(const v8bf*)(pb1 + 16);

    v16bf af  = CAT16(alo, ahi);
    v16bf bf0 = CAT16(b0lo, b0hi);
    v16bf bf1 = CAT16(b1lo, b1hi);
    acc0 = __builtin_amdgcn_wmma_f32_16x16x32_bf16(false, af, false, bf0,
                                                   (short)0, acc0, false, false);
    acc1 = __builtin_amdgcn_wmma_f32_16x16x32_bf16(false, af, false, bf1,
                                                   (short)0, acc1, false, false);
    alo = nalo; ahi = nahi;
    b0lo = nb0lo; b0hi = nb0hi;
    b1lo = nb1lo; b1hi = nb1hi;
  }
  {
    v16bf af  = CAT16(alo, ahi);
    v16bf bf0 = CAT16(b0lo, b0hi);
    v16bf bf1 = CAT16(b1lo, b1hi);
    acc0 = __builtin_amdgcn_wmma_f32_16x16x32_bf16(false, af, false, bf0,
                                                   (short)0, acc0, false, false);
    acc1 = __builtin_amdgcn_wmma_f32_16x16x32_bf16(false, af, false, bf1,
                                                   (short)0, acc1, false, false);
  }

  // C layout: VGPR r -> row m0 + r + 8*(lane>=16), col n0 + (lane&15)
  const int ccol  = n0 + (lane & 15);
  const int rbase = m0 + ((lane >> 4) << 3);
  const float bv0 = bias ? bias[ccol] : 0.0f;
  const float bv1 = bias ? bias[ccol + 16] : 0.0f;
#pragma unroll
  for (int r = 0; r < 8; ++r) {
    float* cr = C + (size_t)(rbase + r) * N + ccol;
    cr[0]  = acc0[r] + bv0;
    cr[16] = acc1[r] + bv1;
  }
}

// ---------------------------- attention ------------------------------------
// One block (256 thr) per batch row b:
//   e[t]  = tanh(lh[b]+ls[b,t]) . Wo + bo      (48 block reductions)
//   att   = softmax_t(e)
//   c[b]  = sum_t att[t] * hiddens[b,t,:]
__global__ void k_attention(const float* __restrict__ lh,
                            const float* __restrict__ ls,
                            const float* __restrict__ hiddens,
                            const float* __restrict__ Wo,
                            const float* __restrict__ bo,
                            float* __restrict__ c,
                            __bf16* __restrict__ c_bf) {
  const int b = blockIdx.x;
  const int tid = threadIdx.x;                    // 256
  __shared__ float red[8];
  __shared__ float att[T_];

  const float lh0 = lh[b * H_ + tid];
  const float lh1 = lh[b * H_ + 256 + tid];
  const float wo0 = Wo[tid], wo1 = Wo[256 + tid];
  const float* lsb = ls + (size_t)b * T_ * H_;

  for (int t = 0; t < T_; ++t) {
    float p = tanhf(lh0 + lsb[t * H_ + tid]) * wo0
            + tanhf(lh1 + lsb[t * H_ + 256 + tid]) * wo1;
#pragma unroll
    for (int off = 16; off > 0; off >>= 1) p += __shfl_xor(p, off, 32);
    if ((tid & 31) == 0) red[tid >> 5] = p;
    __syncthreads();
    if (tid == 0) {
      float s = bo[0];
#pragma unroll
      for (int wv = 0; wv < 8; ++wv) s += red[wv];
      att[t] = s;                                 // raw logits for now
    }
    __syncthreads();
  }

  if (tid == 0) {                                 // 48-wide softmax, serial
    float mx = att[0];
    for (int t = 1; t < T_; ++t) mx = fmaxf(mx, att[t]);
    float s = 0.0f;
    for (int t = 0; t < T_; ++t) { float v = __expf(att[t] - mx); att[t] = v; s += v; }
    float inv = 1.0f / s;
    for (int t = 0; t < T_; ++t) att[t] *= inv;
  }
  __syncthreads();

  const float* hb = hiddens + (size_t)b * T_ * H_;
  float c0 = 0.0f, c1 = 0.0f;
  for (int t = 0; t < T_; ++t) {
    const float a = att[t];
    c0 += a * hb[t * H_ + tid];
    c1 += a * hb[t * H_ + 256 + tid];
  }
  c[b * H_ + tid] = c0;            c[b * H_ + 256 + tid] = c1;
  c_bf[b * H_ + tid] = (__bf16)c0; c_bf[b * H_ + 256 + tid] = (__bf16)c1;
}

// ---------------------------- GRU gates ------------------------------------
// gi = (x[:, :H] @ W_ihS^T + b_ih) precomputed; rank-1 term extra*wlast added
// here so the gi buffer stays reusable across the K look-ahead steps.
__global__ void k_gru_gate(const float* __restrict__ gi,
                           const float* __restrict__ gh,
                           const float* __restrict__ wlast,
                           const float* __restrict__ extra, int estride,
                           float* __restrict__ d,
                           __bf16* __restrict__ d_bf) {
  const int idx = blockIdx.x * blockDim.x + threadIdx.x;   // B*H
  const int b = idx >> 9, h = idx & (H_ - 1);
  const float a = extra[b * estride];
  const size_t base = (size_t)b * G3_;
  const float ir = gi[base + h]           + a * wlast[h];
  const float iz = gi[base + H_ + h]      + a * wlast[H_ + h];
  const float in = gi[base + 2 * H_ + h]  + a * wlast[2 * H_ + h];
  const float hr = gh[base + h];
  const float hz = gh[base + H_ + h];
  const float hn = gh[base + 2 * H_ + h];
  const float r = 1.0f / (1.0f + __expf(-(ir + hr)));
  const float z = 1.0f / (1.0f + __expf(-(iz + hz)));
  const float n = tanhf(in + r * hn);
  const float dn = (1.0f - z) * n + z * d[idx];
  d[idx] = dn;
  d_bf[idx] = (__bf16)dn;
}

// ---------------------------- output MLP -----------------------------------
__device__ __forceinline__ float lrelu(float x) { return x > 0.0f ? x : 0.01f * x; }

// One block (256 thr) per batch row: 512->256->128->64->1, LeakyReLU each.
__global__ void k_mlp(const float* __restrict__ d,
                      const float* __restrict__ W1, const float* __restrict__ b1,
                      const float* __restrict__ W2, const float* __restrict__ b2,
                      const float* __restrict__ W3, const float* __restrict__ b3,
                      const float* __restrict__ W4, const float* __restrict__ b4,
                      float* __restrict__ out, int ostride) {
  const int b = blockIdx.x;
  const int tid = threadIdx.x;
  __shared__ float din[512];
  __shared__ float h1[256];
  __shared__ float h2[128];
  __shared__ float h3[64];
  din[tid] = d[b * H_ + tid];
  din[256 + tid] = d[b * H_ + 256 + tid];
  __syncthreads();
  float s = b1[tid];
  for (int k = 0; k < 512; ++k) s += din[k] * W1[k * 256 + tid];
  h1[tid] = lrelu(s);
  __syncthreads();
  if (tid < 128) {
    float s2 = b2[tid];
    for (int k = 0; k < 256; ++k) s2 += h1[k] * W2[k * 128 + tid];
    h2[tid] = lrelu(s2);
  }
  __syncthreads();
  if (tid < 64) {
    float s3 = b3[tid];
    for (int k = 0; k < 128; ++k) s3 += h2[k] * W3[k * 64 + tid];
    h3[tid] = lrelu(s3);
  }
  __syncthreads();
  if (tid == 0) {
    float s4 = b4[0];
    for (int k = 0; k < 64; ++k) s4 += h3[k] * W4[k];
    out[b * ostride] = lrelu(s4);
  }
}

// ---------------------------- host driver ----------------------------------

extern "C" void kernel_launch(void* const* d_in, const int* in_sizes, int n_in,
                              void* d_out, int out_size, void* d_ws, size_t ws_size,
                              hipStream_t stream) {
  const float* hiddens = (const float*)d_in[0];   // [B,T,S]
  const float* allys   = (const float*)d_in[1];   // [B,T]
  // d_in[2] = k_wk_ahead scalar (fixed = 4)
  const float* Wh  = (const float*)d_in[3];       // [H,S]
  const float* bh  = (const float*)d_in[4];
  const float* Ws  = (const float*)d_in[5];       // [S,S]
  const float* bs  = (const float*)d_in[6];
  const float* Wo  = (const float*)d_in[7];       // [S,1]
  const float* bo  = (const float*)d_in[8];
  const float* Wih = (const float*)d_in[9];       // [3H, S+1]
  const float* Whh = (const float*)d_in[10];      // [3H, H]
  const float* bih = (const float*)d_in[11];
  const float* bhh = (const float*)d_in[12];
  const float* W1 = (const float*)d_in[13]; const float* b1 = (const float*)d_in[14];
  const float* W2 = (const float*)d_in[15]; const float* b2 = (const float*)d_in[16];
  const float* W3 = (const float*)d_in[17]; const float* b3 = (const float*)d_in[18];
  const float* W4 = (const float*)d_in[19]; const float* b4 = (const float*)d_in[20];
  float* out = (float*)d_out;                     // [B, K_AH]

  // workspace carve-up (~47 MB)
  char* w = (char*)d_ws;
  auto carve = [&](size_t bytes) -> void* {
    void* p = (void*)w; w += (bytes + 255) & ~(size_t)255; return p;
  };
  float*  ls     = (float*) carve((size_t)B_ * T_ * H_ * 4);
  __bf16* hid_bf = (__bf16*)carve((size_t)B_ * T_ * H_ * 2);
  __bf16* WsT    = (__bf16*)carve((size_t)H_ * H_ * 2);
  __bf16* WhT    = (__bf16*)carve((size_t)H_ * H_ * 2);
  __bf16* WihS   = (__bf16*)carve((size_t)G3_ * H_ * 2);
  __bf16* WhhB   = (__bf16*)carve((size_t)G3_ * H_ * 2);
  float*  wlast  = (float*) carve((size_t)G3_ * 4);
  float*  dstate = (float*) carve((size_t)B_ * H_ * 4);
  __bf16* d_bf   = (__bf16*)carve((size_t)B_ * H_ * 2);
  float*  lh     = (float*) carve((size_t)B_ * H_ * 4);
  float*  cvec   = (float*) carve((size_t)B_ * H_ * 4);
  __bf16* c_bf   = (__bf16*)carve((size_t)B_ * H_ * 2);
  float*  gi     = (float*) carve((size_t)B_ * G3_ * 4);
  float*  gh     = (float*) carve((size_t)B_ * G3_ * 4);

  // ---- one-time prep (recomputed every call: deterministic) ----
  k_convert_bf16<<<(B_ * T_ * H_ + 255) / 256, 256, 0, stream>>>(hiddens, hid_bf, B_ * T_ * H_);
  k_transpose_bf16<<<(H_ * H_ + 255) / 256, 256, 0, stream>>>(Ws, WsT, H_, H_);
  k_transpose_bf16<<<(H_ * H_ + 255) / 256, 256, 0, stream>>>(Wh, WhT, H_, H_);
  k_split_wih<<<G3_, 256, 0, stream>>>(Wih, WihS, wlast);
  k_convert_bf16<<<(G3_ * H_ + 255) / 256, 256, 0, stream>>>(Whh, WhhB, G3_ * H_);
  k_zero_state<<<(B_ * H_ + 255) / 256, 256, 0, stream>>>(dstate, d_bf, B_ * H_);

  // ls = hiddens @ Ws + bs   (time-invariant, hoisted: M = B*T = 12288)
  k_wmma_gemm<<<dim3((B_ * T_) / 32, H_ / 128), 256, 0, stream>>>(hid_bf, WsT, bs, ls, H_, H_);

  const dim3 gLH(B_ / 32, H_ / 128);   // 256x512 GEMMs
  const dim3 gG (B_ / 32, G3_ / 128);  // 256x1536 GEMMs

  // ---- T sequential decoder steps ----
  for (int t = 0; t < T_; ++t) {
    k_wmma_gemm<<<gLH, 256, 0, stream>>>(d_bf, WhT, bh, lh, H_, H_);
    k_attention<<<B_, 256, 0, stream>>>(lh, ls, hiddens, Wo, bo, cvec, c_bf);
    k_wmma_gemm<<<gG, 256, 0, stream>>>(c_bf, WihS, bih, gi, G3_, H_);
    k_wmma_gemm<<<gG, 256, 0, stream>>>(d_bf, WhhB, bhh, gh, G3_, H_);
    k_gru_gate<<<(B_ * H_) / 256, 256, 0, stream>>>(gi, gh, wlast,
                                                    allys + t, T_, dstate, d_bf);
  }

  // ---- K look-ahead steps ----
  // gi still holds c_last @ W_ihS^T + b_ih from t = T-1 (c_last fixed here),
  // so only gh needs recomputing per step; the scalar feature is out_k.
  for (int k = 0; k < K_AH; ++k) {
    k_mlp<<<B_, 256, 0, stream>>>(dstate, W1, b1, W2, b2, W3, b3, W4, b4,
                                  out + k, K_AH);
    if (k < K_AH - 1) {                           // last GRU result is unused
      k_wmma_gemm<<<gG, 256, 0, stream>>>(d_bf, WhhB, bhh, gh, G3_, H_);
      k_gru_gate<<<(B_ * H_) / 256, 256, 0, stream>>>(gi, gh, wlast,
                                                      out + k, K_AH, dstate, d_bf);
    }
  }
}
